// LDA_47476568490654
// MI455X (gfx1250) — matrix operand order repaired and checked by reading, not compiled, not run on hardware
//
#include <hip/hip_runtime.h>
#include <stdint.h>

// ---------------------------------------------------------------------------
// out[b,d,n] = sum_m adj[b,n,m] * x[b,d,m]   (B=2, D=3072, N=8192, fp32)
// Per batch: OUT = X * ADJ^T (both operands K-major) -> WMMA NT GEMM.
// Compute-bound (AI ~ 880 FLOP/B). bf16x3 split scheme (hi/lo planes) with
// v_wmma_f32_16x16x32_bf16 + fp32 accumulate: ~16 mantissa bits at ~2.7x the
// fp32-WMMA (16x16x4) rate.
//
// Fast path (needs ~704 MB workspace): one-time fp32 -> {hi,lo} bf16 plane
// split in global memory, then the GEMM streams raw bf16 tiles into LDS with
// the Tensor Data Mover (tensor_load_to_lds, TENSORcnt, double-buffered LDS).
// Fallback path: round-1 fused kernel (convert while staging through VGPRs).
// ---------------------------------------------------------------------------

typedef __attribute__((ext_vector_type(16))) __bf16 v16bf;
typedef __attribute__((ext_vector_type(8)))  __bf16 v8bf;
typedef __attribute__((ext_vector_type(4)))  __bf16 v4bf;
typedef __attribute__((ext_vector_type(8)))  float  v8f;
typedef __attribute__((ext_vector_type(4)))  float  f4;
typedef __attribute__((ext_vector_type(4)))  unsigned int u32x4;
typedef __attribute__((ext_vector_type(8)))  int    i32x8;
typedef __attribute__((ext_vector_type(4)))  int    i32x4;

#define BM 128          // output rows (d) per block
#define BN 128          // output cols (n) per block
#define BK 64           // k (m) staged per slab
#define LSTR 72         // LDS row stride (bf16): 144 B = 128 B data + 16 B pad
#define PLANE_HALVES (BM * LSTR)         // 9216 bf16
#define PLANE_BYTES  (PLANE_HALVES * 2)  // 18432 B
#define BUF_BYTES    (4 * PLANE_BYTES)   // 73728 B  (Xh, Xl, Ah, Al)

#if defined(__HIP_DEVICE_COMPILE__) && \
    __has_builtin(__builtin_amdgcn_tensor_load_to_lds) && \
    __has_builtin(__builtin_amdgcn_s_wait_tensorcnt)
#define HAVE_TDM 1
#else
#define HAVE_TDM 0
#endif

__device__ __forceinline__ v16bf cat16(v8bf a, v8bf b) {
  return __builtin_shufflevector(a, b, 0,1,2,3,4,5,6,7,8,9,10,11,12,13,14,15);
}

__device__ __forceinline__ v8f wmma_bf16(v16bf a, v16bf b, v8f c) {
  return __builtin_amdgcn_wmma_f32_16x16x32_bf16(false, a, false, b, (short)0, c,
                                                 false, false);
}

// One BK=64 slab of bf16x3 MMAs from 4 LDS planes (layouts per ISA 7.12.2).
__device__ __forceinline__ void mma_slab(const __bf16* pXh, const __bf16* pXl,
                                         const __bf16* pAh, const __bf16* pAl,
                                         int wm, int wn, int hs, int r16,
                                         v8f (&acc)[2][4]) {
#pragma unroll
  for (int kc = 0; kc < BK; kc += 32) {
    v16bf fah[2], fal[2], fbh[4], fbl[4];
    // A (16x32): elem e<8 -> K=8*hs+e ; e>=8 -> K=16+8*hs+(e-8)
#pragma unroll
    for (int i = 0; i < 2; ++i) {
      const int r = wm * 32 + i * 16 + r16;
      const __bf16* ph = pXh + r * LSTR + kc + 8 * hs;
      const __bf16* pl = pXl + r * LSTR + kc + 8 * hs;
      fah[i] = cat16(*(const v8bf*)ph, *(const v8bf*)(ph + 16));
      fal[i] = cat16(*(const v8bf*)pl, *(const v8bf*)(pl + 16));
    }
    // B (32x16): B[k][n]=adj[n][k]; elem e<8 -> K=8*hs+e ; e>=8 -> K=16+8*hs+(e-8)
#pragma unroll
    for (int j = 0; j < 4; ++j) {
      const int r = wn * 64 + j * 16 + r16;
      const __bf16* ph = pAh + r * LSTR + kc + 8 * hs;
      const __bf16* pl = pAl + r * LSTR + kc + 8 * hs;
      fbh[j] = cat16(*(const v8bf*)ph, *(const v8bf*)(ph + 16));
      fbl[j] = cat16(*(const v8bf*)pl, *(const v8bf*)(pl + 16));
    }
    // hi*hi + lo*hi + hi*lo (drop lo*lo ~ 2^-16)
#pragma unroll
    for (int i = 0; i < 2; ++i)
#pragma unroll
      for (int j = 0; j < 4; ++j) {
        acc[i][j] = wmma_bf16(fah[i], fbh[j], acc[i][j]);
        acc[i][j] = wmma_bf16(fal[i], fbh[j], acc[i][j]);
        acc[i][j] = wmma_bf16(fah[i], fbl[j], acc[i][j]);
      }
  }
}

// ---------------------------------------------------------------------------
// TDM: one descriptor-driven DMA per 128x64 bf16 tile, global -> LDS, with
// hardware row padding (32 DWORDs data + 4 DWORDs pad) == LSTR layout.
// clang-23 toolchain form: (u32x4 g0, i32x8 g1, i32x4 g2, i32x4 g3, i32x8, cpol)
// ---------------------------------------------------------------------------
__device__ __forceinline__ void tdm_tile(const __bf16* plane, size_t row0,
                                         int m0, int Nw, unsigned lds_off) {
#if HAVE_TDM
  const unsigned long long ga =
      (unsigned long long)(uintptr_t)plane + ((row0 * (size_t)Nw + (size_t)m0) * 2ull);
  u32x4 g0;
  g0[0] = 1u;                                    // count=1, user descriptor
  g0[1] = lds_off;                               // lds_addr (bytes)
  g0[2] = (unsigned)(ga & 0xFFFFFFFFu);          // global_addr[31:0]
  g0[3] = (unsigned)((ga >> 32) & 0x01FFFFFFu)   // global_addr[56:32]
        | 0x80000000u;                           // type=2 ("image")
  const unsigned td0 = (unsigned)Nw;             // tensor_dim0 (never OOB here)
  const unsigned td1 = 1u << 24;                 // tensor_dim1 (never OOB here)
  i32x8 g1;
  g1[0] = 0x07110000;                            // data_size=2B, pad_enable,
                                                 // pad_interval=32dw, pad_amount=4dw
  g1[1] = (int)((td0 & 0xFFFFu) << 16);          // abar=0 | tensor_dim0 lo
  g1[2] = (int)(((td0 >> 16) & 0xFFFFu) | ((td1 & 0xFFFFu) << 16));
  g1[3] = (int)(((td1 >> 16) & 0xFFFFu) | (64u << 16));   // tile_dim0 = 64
  g1[4] = 128;                                   // tile_dim1=128, tile_dim2=0
  g1[5] = Nw;                                    // tensor_dim0_stride[31:0] = N
  g1[6] = 0;                                     // stride hi | dim1_stride lo
  g1[7] = 0;
  const i32x4 gz4 = {0, 0, 0, 0};                // groups 2/3: dims 2..4 unused
  const i32x8 gz8 = {0, 0, 0, 0, 0, 0, 0, 0};
  __builtin_amdgcn_tensor_load_to_lds(g0, g1, gz4, gz4, gz8, 0);
#else
  (void)plane; (void)row0; (void)m0; (void)Nw; (void)lds_off;
#endif
}

// ---------------------------------------------------------------------------
// Pre-split: fp32 -> bf16 hi/lo planes (one pass over x and adj).
// ---------------------------------------------------------------------------
__global__ __launch_bounds__(256) void split_bf16_kernel(
    const float* __restrict__ src, __bf16* __restrict__ hi,
    __bf16* __restrict__ lo, size_t count4) {
  size_t i = (size_t)blockIdx.x * blockDim.x + threadIdx.x;
  const size_t stride = (size_t)gridDim.x * blockDim.x;
  for (; i < count4; i += stride) {
    f4 v = ((const f4*)src)[i];
    v4bf h, l;
#pragma unroll
    for (int c = 0; c < 4; ++c) {
      __bf16 hh = (__bf16)v[c];
      h[c] = hh;
      l[c] = (__bf16)(v[c] - (float)hh);
    }
    ((v4bf*)hi)[i] = h;
    ((v4bf*)lo)[i] = l;
  }
}

// ---------------------------------------------------------------------------
// Fast GEMM: bf16 planes streamed via TDM, double-buffered LDS.
// ---------------------------------------------------------------------------
__global__ __launch_bounds__(256) void gcn_bmm_planes_kernel(
    const __bf16* __restrict__ XH, const __bf16* __restrict__ XL,
    const __bf16* __restrict__ AH, const __bf16* __restrict__ AL,
    float* __restrict__ OUT, int D, int N) {
  extern __shared__ __bf16 smem[];  // 2 x BUF_BYTES (dynamic LDS base = offset 0)

  const int tid  = threadIdx.x;
  const int lane = tid & 31;
  const int wid  = tid >> 5;
  const int wm   = wid >> 1;
  const int wn   = wid & 1;
  const int hs   = lane >> 4;
  const int r16  = lane & 15;

  const int d0 = blockIdx.x * BM;
  const int n0 = blockIdx.y * BN;
  const int b  = blockIdx.z;
  const size_t xrow0 = (size_t)b * D + d0;   // row base in [B*D, N] x-planes
  const size_t arow0 = (size_t)b * N + n0;   // row base in [B*N, N] adj-planes
  float* ob = OUT + (size_t)b * D * N;

  v8f acc[2][4];
  {
    v8f z = {};
#pragma unroll
    for (int i = 0; i < 2; ++i)
#pragma unroll
      for (int j = 0; j < 4; ++j) acc[i][j] = z;
  }

  const int nslab = N / BK;

#if HAVE_TDM
  if (wid == 0) {  // TDM ignores EXEC; exactly one wave issues the DMAs
    tdm_tile(XH, xrow0, 0, N, 0 * PLANE_BYTES);
    tdm_tile(XL, xrow0, 0, N, 1 * PLANE_BYTES);
    tdm_tile(AH, arow0, 0, N, 2 * PLANE_BYTES);
    tdm_tile(AL, arow0, 0, N, 3 * PLANE_BYTES);
  }
  for (int s = 0; s < nslab; ++s) {
    const int cur = s & 1;
    if (wid == 0) __builtin_amdgcn_s_wait_tensorcnt(0);  // slab s landed
    __syncthreads();                                     // publish to all waves
    if (wid == 0 && s + 1 < nslab) {                     // prefetch next slab
      const int m1 = (s + 1) * BK;
      const unsigned boff = (unsigned)((cur ^ 1) * BUF_BYTES);
      tdm_tile(XH, xrow0, m1, N, boff + 0 * PLANE_BYTES);
      tdm_tile(XL, xrow0, m1, N, boff + 1 * PLANE_BYTES);
      tdm_tile(AH, arow0, m1, N, boff + 2 * PLANE_BYTES);
      tdm_tile(AL, arow0, m1, N, boff + 3 * PLANE_BYTES);
    }
    const __bf16* base = smem + cur * (BUF_BYTES / 2);
    mma_slab(base, base + PLANE_HALVES, base + 2 * PLANE_HALVES,
             base + 3 * PLANE_HALVES, wm, wn, hs, r16, acc);
    __syncthreads();  // all reads of buf done before slab s+2 overwrites it
  }
#else
  // Fallback staging: plain coalesced bf16 copies through VGPRs.
  const int crow = tid >> 1;         // 0..127
  const int coff = (tid & 1) * 32;   // bf16 elems
  for (int s = 0; s < nslab; ++s) {
    const int m0 = s * BK;
    __syncthreads();
    const __bf16* gsrc[4] = {XH + (xrow0 + crow) * (size_t)N,
                             XL + (xrow0 + crow) * (size_t)N,
                             AH + (arow0 + crow) * (size_t)N,
                             AL + (arow0 + crow) * (size_t)N};
#pragma unroll
    for (int p = 0; p < 4; ++p) {
      const __bf16* g = gsrc[p] + m0 + coff;
      __bf16* l = smem + p * PLANE_HALVES + crow * LSTR + coff;
#pragma unroll
      for (int q = 0; q < 4; ++q)
        *(v8bf*)(l + 8 * q) = *(const v8bf*)(g + 8 * q);
    }
    __syncthreads();
    mma_slab(smem, smem + PLANE_HALVES, smem + 2 * PLANE_HALVES,
             smem + 3 * PLANE_HALVES, wm, wn, hs, r16, acc);
  }
#endif

  // D layout: VGPR v, lanes 0-15 -> M=v, lanes 16-31 -> M=v+8
#pragma unroll
  for (int i = 0; i < 2; ++i) {
#pragma unroll
    for (int j = 0; j < 4; ++j) {
      const int n  = n0 + wn * 64 + j * 16 + r16;
      const int dr = d0 + wm * 32 + i * 16 + 8 * hs;
#pragma unroll
      for (int v = 0; v < 8; ++v)
        ob[(size_t)(dr + v) * N + n] = acc[i][j][v];
    }
  }
}

// ---------------------------------------------------------------------------
// Fallback fused kernel (round-1): split fp32 -> hi/lo while staging.
// ---------------------------------------------------------------------------
__global__ __launch_bounds__(256) void gcn_bmm_fused_kernel(
    const float* __restrict__ X, const float* __restrict__ ADJ,
    float* __restrict__ OUT, int D, int N) {
  extern __shared__ __bf16 smem[];
  __bf16* sXh = smem;
  __bf16* sXl = smem + 1 * PLANE_HALVES;
  __bf16* sAh = smem + 2 * PLANE_HALVES;
  __bf16* sAl = smem + 3 * PLANE_HALVES;

  const int tid  = threadIdx.x;
  const int lane = tid & 31;
  const int wid  = tid >> 5;
  const int wm   = wid >> 1;
  const int wn   = wid & 1;
  const int hs   = lane >> 4;
  const int r16  = lane & 15;

  const int d0 = blockIdx.x * BM;
  const int n0 = blockIdx.y * BN;
  const int b  = blockIdx.z;

  const float* xb = X   + (size_t)b * D * N;
  const float* ab = ADJ + (size_t)b * N * N;
  float*       ob = OUT + (size_t)b * D * N;

  const int lrow = tid >> 4;
  const int lcol = (tid & 15) * 4;

  v8f acc[2][4];
  {
    v8f z = {};
#pragma unroll
    for (int i = 0; i < 2; ++i)
#pragma unroll
      for (int j = 0; j < 4; ++j) acc[i][j] = z;
  }

  for (int m0 = 0; m0 < N; m0 += BK) {
    __syncthreads();
#pragma unroll
    for (int r = 0; r < BM; r += 16) {
      const int row = r + lrow;
      f4 vx = *(const f4*)(xb + (size_t)(d0 + row) * N + m0 + lcol);
      f4 va = *(const f4*)(ab + (size_t)(n0 + row) * N + m0 + lcol);
      v4bf xh, xl, ah, al;
#pragma unroll
      for (int c = 0; c < 4; ++c) {
        __bf16 h;
        h = (__bf16)vx[c]; xh[c] = h; xl[c] = (__bf16)(vx[c] - (float)h);
        h = (__bf16)va[c]; ah[c] = h; al[c] = (__bf16)(va[c] - (float)h);
      }
      *(v4bf*)(sXh + row * LSTR + lcol) = xh;
      *(v4bf*)(sXl + row * LSTR + lcol) = xl;
      *(v4bf*)(sAh + row * LSTR + lcol) = ah;
      *(v4bf*)(sAl + row * LSTR + lcol) = al;
    }
    __syncthreads();
    mma_slab(sXh, sXl, sAh, sAl, wm, wn, hs, r16, acc);
  }

#pragma unroll
  for (int i = 0; i < 2; ++i) {
#pragma unroll
    for (int j = 0; j < 4; ++j) {
      const int n  = n0 + wn * 64 + j * 16 + r16;
      const int dr = d0 + wm * 32 + i * 16 + 8 * hs;
#pragma unroll
      for (int v = 0; v < 8; ++v)
        ob[(size_t)(dr + v) * N + n] = acc[i][j][v];
    }
  }
}

extern "C" void kernel_launch(void* const* d_in, const int* in_sizes, int n_in,
                              void* d_out, int out_size, void* d_ws, size_t ws_size,
                              hipStream_t stream) {
  (void)in_sizes; (void)n_in; (void)out_size;
  constexpr int B = 2, D = 3072, N = 8192;
  const size_t SZX = (size_t)B * D * N;   // x elements
  const size_t SZA = (size_t)B * N * N;   // adj elements
  const size_t need = (SZX + SZA) * 2 * sizeof(__bf16);  // hi+lo planes (~704 MB)

  const float* x   = (const float*)d_in[0];
  const float* adj = (const float*)d_in[1];
  float* out = (float*)d_out;

  dim3 grid(D / BM, N / BN, B);  // x = d-tile fastest: neighbors share adj slab in L2
  dim3 block(256);

  if (d_ws != nullptr && ws_size >= need) {
    __bf16* xh = (__bf16*)d_ws;
    __bf16* xl = xh + SZX;
    __bf16* ah = xl + SZX;
    __bf16* al = ah + SZA;
    split_bf16_kernel<<<2048, block, 0, stream>>>(x, xh, xl, SZX / 4);
    split_bf16_kernel<<<4096, block, 0, stream>>>(adj, ah, al, SZA / 4);
    gcn_bmm_planes_kernel<<<grid, block, 2 * BUF_BYTES, stream>>>(xh, xl, ah, al,
                                                                  out, D, N);
  } else {
    gcn_bmm_fused_kernel<<<grid, block, BUF_BYTES, stream>>>(x, adj, out, D, N);
  }
}
